// HilbertAwareMultiScaleAttention_43112881717571
// MI455X (gfx1250) — matrix operand order repaired and checked by reading, not compiled, not run on hardware
//
#include <hip/hip_runtime.h>
#include <hip/hip_bf16.h>
#include <math.h>

static constexpr int SEQ    = 1536;
static constexpr int DMODEL = 1024;
static constexpr int NH     = 16;
static constexpr int DHC    = 64;
static constexpr int NKB    = SEQ / 32;   // 48 key blocks of 32

typedef _Float16 v16h __attribute__((ext_vector_type(16)));
typedef _Float16 v8h  __attribute__((ext_vector_type(8)));
typedef float    v8f  __attribute__((ext_vector_type(8)));
typedef float    v4f  __attribute__((ext_vector_type(4)));

__device__ __forceinline__ v8f wmma16(v16h a, v16h b, v8f c) {
  return __builtin_amdgcn_wmma_f32_16x16x32_f16(false, a, false, b, (short)0, c, false, false);
}

// A-operand (16x32 f16): lane holds row=lane%16, k = kb..kb+15, kb = 16*(lane/16)
__device__ __forceinline__ v16h load_a_f16(const _Float16* row, int kb2) {
  v8h lo = *(const v8h*)(row + 2 * kb2);
  v8h hi = *(const v8h*)(row + 2 * kb2 + 8);
  v16h r;
#pragma unroll
  for (int i = 0; i < 8; ++i) { r[i] = lo[i]; r[i + 8] = hi[i]; }
  return r;
}

// ---------------- K0: weight transpose + f32->f16 ----------------
__global__ void prep_weights_kernel(const float* __restrict__ wqkv,
                                    const float* __restrict__ wout,
                                    _Float16* __restrict__ wqkvT,
                                    _Float16* __restrict__ woutT) {
  int idx = blockIdx.x * blockDim.x + threadIdx.x;
  const int T1 = 3072 * 1024;
  const int T2 = 1024 * 1024;
  if (idx < T1) {
    int n = idx >> 10, k = idx & 1023;
    wqkvT[idx] = (_Float16)wqkv[k * 3072 + n];
  } else if (idx < T1 + T2) {
    int j = idx - T1;
    int n = j >> 10, k = j & 1023;
    woutT[j] = (_Float16)wout[k * 1024 + n];
  }
}

// ---------------- K1: layernorm + per-token stats ----------------
__global__ void ln_stats_kernel(const float* __restrict__ x,
                                const int* __restrict__ levels,
                                const float* __restrict__ ln_g,
                                const float* __restrict__ ln_b,
                                const float* __restrict__ level_emb,
                                const float* __restrict__ scale_w,
                                _Float16* __restrict__ xnh,
                                float* __restrict__ coef,
                                float* __restrict__ pathsf,
                                float* __restrict__ pathnorm,
                                int* __restrict__ depthbuf) {
  int i = blockIdx.x;
  int tid = threadIdx.x;
  __shared__ float red0[8], red1[8];
  __shared__ float smu, srstd;
  const float* xr = x + (size_t)i * DMODEL;
  float s = 0.f, s2 = 0.f;
#pragma unroll
  for (int e = 0; e < 4; ++e) { float v = xr[tid + 256 * e]; s += v; s2 += v * v; }
#pragma unroll
  for (int off = 16; off > 0; off >>= 1) {
    s  += __shfl_xor(s,  off, 32);
    s2 += __shfl_xor(s2, off, 32);
  }
  int wid = tid >> 5;
  if ((tid & 31) == 0) { red0[wid] = s; red1[wid] = s2; }
  __syncthreads();
  if (tid == 0) {
    float ts = 0.f, ts2 = 0.f;
    for (int w = 0; w < 8; ++w) { ts += red0[w]; ts2 += red1[w]; }
    float mu = ts / (float)DMODEL;
    float var = ts2 / (float)DMODEL - mu * mu;
    smu = mu;
    srstd = rsqrtf(var + 1e-5f);
  }
  __syncthreads();
  float mu = smu, rstd = srstd;
#pragma unroll
  for (int e = 0; e < 4; ++e) {
    int c = tid + 256 * e;
    float v = (xr[c] - mu) * rstd * ln_g[c] + ln_b[c];
    xnh[(size_t)i * DMODEL + c] = (_Float16)v;
  }
  if (tid < 16) {
    int dep = levels[i * 9];
    int dc = dep < 0 ? 0 : (dep > 50 ? 50 : dep);
    coef[tid * SEQ + i] = 0.125f * scale_w[tid] * level_emb[dc * 16 + tid];
  }
  if (tid == 16) {
    float nn = 0.f;
    for (int p = 0; p < 8; ++p) {
      float pf = (float)levels[i * 9 + 1 + p];
      pathsf[i * 8 + p] = pf;
      nn += pf * pf;
    }
    pathnorm[i] = fmaxf(sqrtf(nn), 1e-8f);
    depthbuf[i] = levels[i * 9];
  }
}

// ---------------- K2: QKV GEMM (xn @ w_qkv), 16x64 tile per wave ----------------
__global__ void qkv_gemm_kernel(const _Float16* __restrict__ xnh,
                                const _Float16* __restrict__ wqkvT,
                                _Float16* __restrict__ qb,
                                _Float16* __restrict__ kbuf,
                                _Float16* __restrict__ vT) {
  int n0 = blockIdx.x * 64, m0 = blockIdx.y * 16;
  int lane = threadIdx.x, lh = lane >> 4, lr = lane & 15;
  int kb2 = lh * 8;
  const _Float16* arow  = xnh + (size_t)(m0 + lr) * DMODEL;
  const _Float16* brow0 = wqkvT + (size_t)(n0 + lr) * DMODEL + 16 * lh;
  v8f z = {0.f,0.f,0.f,0.f,0.f,0.f,0.f,0.f};
  v8f acc0 = z, acc1 = z, acc2 = z, acc3 = z;

  // rotate-register software pipeline: next iteration's loads overlap this
  // iteration's WMMAs, so their lifetimes cannot be coalesced away.
  v16h a  = load_a_f16(arow, kb2);
  v16h b0 = *(const v16h*)(brow0 + (size_t)0 * 16 * DMODEL);
  v16h b1 = *(const v16h*)(brow0 + (size_t)1 * 16 * DMODEL);
  v16h b2 = *(const v16h*)(brow0 + (size_t)2 * 16 * DMODEL);
  v16h b3 = *(const v16h*)(brow0 + (size_t)3 * 16 * DMODEL);
#pragma unroll 2
  for (int ks = 0; ks < 32; ++ks) {
    int k0n = ((ks + 1) & 31) * 32;   // last iter harmlessly reloads k0=0
    v16h an  = load_a_f16(arow + k0n, kb2);
    v16h b0n = *(const v16h*)(brow0 + (size_t)0 * 16 * DMODEL + k0n);
    v16h b1n = *(const v16h*)(brow0 + (size_t)1 * 16 * DMODEL + k0n);
    v16h b2n = *(const v16h*)(brow0 + (size_t)2 * 16 * DMODEL + k0n);
    v16h b3n = *(const v16h*)(brow0 + (size_t)3 * 16 * DMODEL + k0n);
    acc0 = wmma16(a, b0, acc0);
    acc1 = wmma16(a, b1, acc1);
    acc2 = wmma16(a, b2, acc2);
    acc3 = wmma16(a, b3, acc3);
    a = an; b0 = b0n; b1 = b1n; b2 = b2n; b3 = b3n;
  }
  v8f accs[4] = {acc0, acc1, acc2, acc3};
  // 64-column tile never straddles q/k/v boundaries: branch is block-uniform.
  if (n0 < 1024) {
    int h = n0 >> 6;
    _Float16* base = qb + ((size_t)(h * SEQ + m0 + 8 * lh)) * DHC;
#pragma unroll
    for (int nt = 0; nt < 4; ++nt) {
      int d = nt * 16 + lr;
#pragma unroll
      for (int r = 0; r < 8; ++r)
        base[(size_t)r * DHC + d] = (_Float16)accs[nt][r];
    }
  } else if (n0 < 2048) {
    int h = (n0 - 1024) >> 6;
    _Float16* base = kbuf + ((size_t)(h * SEQ + m0 + 8 * lh)) * DHC;
#pragma unroll
    for (int nt = 0; nt < 4; ++nt) {
      int d = nt * 16 + lr;
#pragma unroll
      for (int r = 0; r < 8; ++r)
        base[(size_t)r * DHC + d] = (_Float16)accs[nt][r];
    }
  } else {
    int h = (n0 - 2048) >> 6;
#pragma unroll
    for (int nt = 0; nt < 4; ++nt) {
      int d = nt * 16 + lr;
      _Float16* dst = vT + ((size_t)(h * DHC + d)) * SEQ + m0 + 8 * lh;
      v8h pk;
#pragma unroll
      for (int r = 0; r < 8; ++r) pk[r] = (_Float16)accs[nt][r];
      *(v8h*)dst = pk;   // one 16B packed store per d-tile
    }
  }
}

// ---------------- K3: flash-style attention with biases ----------------
__global__ void attn_kernel(const _Float16* __restrict__ qb,
                            const _Float16* __restrict__ kbuf,
                            const _Float16* __restrict__ vT,
                            const float* __restrict__ coef,
                            const float* __restrict__ pathsf,
                            const float* __restrict__ pathnorm,
                            const int* __restrict__ depthbuf,
                            const float* __restrict__ hb_w1,
                            const float* __restrict__ hb_b1,
                            const float* __restrict__ hb_w2,
                            const float* __restrict__ hb_b2,
                            const float* __restrict__ rel_emb,
                            const unsigned char* __restrict__ mask,
                            _Float16* __restrict__ ctxb) {
  int qt = blockIdx.x, h = blockIdx.y;
  int i0 = qt * 16;
  int lane = threadIdx.x, lh = lane >> 4, lr = lane & 15;
  int kb2 = lh * 8;

  __shared__ float s_rel[101 * 16];
  __shared__ float s_w1a[64], s_w1b[64], s_b1[64], s_w2h[64];
  __shared__ float s_qpath[16 * 8], s_kpath[32 * 8];
  __shared__ float s_qn[16], s_kn[32];
  __shared__ int   s_qdep[16], s_kdep[32];
  __shared__ __align__(32) float s_P[16 * 32];

  for (int t = lane; t < 101 * 16; t += 32) s_rel[t] = rel_emb[t];
  for (int t = lane; t < 64; t += 32) {
    s_w1a[t] = hb_w1[t];
    s_w1b[t] = hb_w1[64 + t];
    s_b1[t]  = hb_b1[t];
    s_w2h[t] = hb_w2[t * 16 + h];
  }
  if (lane < 16) {
    int i = i0 + lane;
#pragma unroll
    for (int p = 0; p < 8; ++p) s_qpath[lane * 8 + p] = pathsf[i * 8 + p];
    s_qn[lane] = pathnorm[i];
    s_qdep[lane] = depthbuf[i];
  }
  float b2h = hb_b2[h];
  float coefr[8];
#pragma unroll
  for (int r = 0; r < 8; ++r) coefr[r] = coef[h * SEQ + i0 + r + 8 * lh];

  const _Float16* qrow = qb + ((size_t)(h * SEQ + i0 + lr)) * DHC;
  v16h aq0 = load_a_f16(qrow, kb2);
  v16h aq1 = load_a_f16(qrow + 32, kb2);

  float m[8], l[8];
  v8f c0 = {0.f,0.f,0.f,0.f,0.f,0.f,0.f,0.f};
  v8f c1 = c0, c2 = c0, c3 = c0;
#pragma unroll
  for (int r = 0; r < 8; ++r) { m[r] = -3.402823466e38f; l[r] = 0.f; }
  __syncthreads();

  for (int kblk = 0; kblk < NKB; ++kblk) {
    int j0 = kblk * 32;
    __syncthreads();
    {
      int j = j0 + lane;
#pragma unroll
      for (int p = 0; p < 8; ++p) s_kpath[lane * 8 + p] = pathsf[j * 8 + p];
      s_kn[lane] = pathnorm[j];
      s_kdep[lane] = depthbuf[j];
    }
    __syncthreads();

    // ---- QK^T : two 16x16 score tiles over 32 keys ----
    const _Float16* kr0 = kbuf + ((size_t)(h * SEQ + j0 + lr)) * DHC + 16 * lh;
    const _Float16* kr1 = kbuf + ((size_t)(h * SEQ + j0 + 16 + lr)) * DHC + 16 * lh;
    const _Float16* vbase = vT + ((size_t)(h * DHC)) * SEQ + j0 + 16 * lh;

    // prefetch next key-block operands while the MLP VALU loop runs
    if (kblk + 1 < NKB) {
      __builtin_prefetch((const void*)(kr0 + 32 * DHC), 0, 0);
      __builtin_prefetch((const void*)(kr1 + 32 * DHC), 0, 0);
      __builtin_prefetch((const void*)(vbase + 32), 0, 0);
      __builtin_prefetch((const void*)(vbase + (size_t)(16 + lr) * SEQ + 32), 0, 0);
    }

    v8f z = {0.f,0.f,0.f,0.f,0.f,0.f,0.f,0.f};
    v16h kb00 = *(const v16h*)kr0;
    v16h kb01 = *(const v16h*)(kr0 + 32);
    v16h kb10 = *(const v16h*)kr1;
    v16h kb11 = *(const v16h*)(kr1 + 32);
    v8f s0 = wmma16(aq0, kb00, z);
    v8f s1 = wmma16(aq0, kb10, z);
    s0 = wmma16(aq1, kb01, s0);
    s1 = wmma16(aq1, kb11, s1);

    // ---- per-pair hilbert bias MLP + rel bias + mask ----
    float dist[16], sim[16], acc2[16];
#pragma unroll
    for (int t = 0; t < 2; ++t) {
#pragma unroll
      for (int r = 0; r < 8; ++r) {
        int e = t * 8 + r;
        int qrw = r + 8 * lh;
        int jl = lr + 16 * t;
        float d2 = 0.f, dot = 0.f;
#pragma unroll
        for (int p = 0; p < 8; ++p) {
          float qp = s_qpath[qrw * 8 + p];
          float kp = s_kpath[jl * 8 + p];
          float df = qp - kp;
          d2 = fmaf(df, df, d2);
          dot = fmaf(qp, kp, dot);
        }
        dist[e] = sqrtf(d2);
        sim[e] = dot / (s_qn[qrw] * s_kn[jl]);
        acc2[e] = 0.f;
      }
    }
    for (int k = 0; k < 64; ++k) {
      float wa = s_w1a[k], wb = s_w1b[k], bb = s_b1[k], w2 = s_w2h[k];
#pragma unroll
      for (int e = 0; e < 16; ++e) {
        float hv = fmaf(dist[e], wa, fmaf(sim[e], wb, bb));
        hv = fmaxf(hv, 0.f);
        acc2[e] = fmaf(hv, w2, acc2[e]);
      }
    }
#pragma unroll
    for (int t = 0; t < 2; ++t) {
#pragma unroll
      for (int r = 0; r < 8; ++r) {
        int e = t * 8 + r;
        int irow = i0 + r + 8 * lh;
        int jl = lr + 16 * t;
        int j = j0 + jl;
        float sc = (t ? s1[r] : s0[r]) * coefr[r];
        float hb = tanhf(acc2[e] + b2h);
        if (irow == j) hb = 0.f;
        int ld = s_kdep[jl] - s_qdep[r + 8 * lh];
        ld = ld < -50 ? -50 : (ld > 50 ? 50 : ld);
        sc += 0.1f * hb + 0.05f * s_rel[(ld + 50) * 16 + h];
        if (!mask[(size_t)irow * SEQ + j]) sc = -3.402823466e38f;
        if (t) s1[r] = sc; else s0[r] = sc;
      }
    }

    // ---- online softmax over the 32 new keys ----
#pragma unroll
    for (int r = 0; r < 8; ++r) {
      float mx = fmaxf(s0[r], s1[r]);
      mx = fmaxf(mx, __shfl_xor(mx, 1, 32));
      mx = fmaxf(mx, __shfl_xor(mx, 2, 32));
      mx = fmaxf(mx, __shfl_xor(mx, 4, 32));
      mx = fmaxf(mx, __shfl_xor(mx, 8, 32));
      float mnew = fmaxf(m[r], mx);
      float alpha = __expf(m[r] - mnew);
      float p0 = __expf(s0[r] - mnew);
      float p1 = __expf(s1[r] - mnew);
      float rs = p0 + p1;
      rs += __shfl_xor(rs, 1, 32);
      rs += __shfl_xor(rs, 2, 32);
      rs += __shfl_xor(rs, 4, 32);
      rs += __shfl_xor(rs, 8, 32);
      l[r] = l[r] * alpha + rs;
      m[r] = mnew;
      c0[r] *= alpha; c1[r] *= alpha; c2[r] *= alpha; c3[r] *= alpha;
      s_P[(r + 8 * lh) * 32 + lr]      = p0;
      s_P[(r + 8 * lh) * 32 + lr + 16] = p1;
    }
    __syncthreads();

    // re-layout P (D-layout -> A-operand) through LDS
    v16h pa;
    {
      const float* pr = &s_P[lr * 32 + 2 * kb2];
      v4f f0 = *(const v4f*)pr;
      v4f f1 = *(const v4f*)(pr + 4);
      v4f f2 = *(const v4f*)(pr + 8);
      v4f f3 = *(const v4f*)(pr + 12);
#pragma unroll
      for (int q = 0; q < 4; ++q) {
        pa[q]      = (_Float16)f0[q];
        pa[4 + q]  = (_Float16)f1[q];
        pa[8 + q]  = (_Float16)f2[q];
        pa[12 + q] = (_Float16)f3[q];
      }
    }

    // ---- attn @ V : four 16x16 dh tiles, K = 32 keys ----
    v16h vb0 = *(const v16h*)(vbase + (size_t)(0 * 16 + lr) * SEQ);
    v16h vb1 = *(const v16h*)(vbase + (size_t)(1 * 16 + lr) * SEQ);
    v16h vb2 = *(const v16h*)(vbase + (size_t)(2 * 16 + lr) * SEQ);
    v16h vb3 = *(const v16h*)(vbase + (size_t)(3 * 16 + lr) * SEQ);
    c0 = wmma16(pa, vb0, c0);
    c1 = wmma16(pa, vb1, c1);
    c2 = wmma16(pa, vb2, c2);
    c3 = wmma16(pa, vb3, c3);
  }

  // normalize and write ctx (f16, [s][1024])
#pragma unroll
  for (int r = 0; r < 8; ++r) {
    float inv = 1.0f / l[r];
    size_t orow = (size_t)(i0 + r + 8 * lh) * DMODEL + h * DHC + lr;
    ctxb[orow + 0]  = (_Float16)(c0[r] * inv);
    ctxb[orow + 16] = (_Float16)(c1[r] * inv);
    ctxb[orow + 32] = (_Float16)(c2[r] * inv);
    ctxb[orow + 48] = (_Float16)(c3[r] * inv);
  }
}

// ---------------- K4: output projection, 16x64 tile per wave ----------------
__global__ void out_gemm_kernel(const _Float16* __restrict__ ctxb,
                                const _Float16* __restrict__ woutT,
                                const float* __restrict__ b_out,
                                float* __restrict__ out) {
  int n0 = blockIdx.x * 64, m0 = blockIdx.y * 16;
  int lane = threadIdx.x, lh = lane >> 4, lr = lane & 15;
  int kb2 = lh * 8;
  const _Float16* arow  = ctxb + (size_t)(m0 + lr) * DMODEL;
  const _Float16* brow0 = woutT + (size_t)(n0 + lr) * DMODEL + 16 * lh;
  v8f z = {0.f,0.f,0.f,0.f,0.f,0.f,0.f,0.f};
  v8f acc0 = z, acc1 = z, acc2 = z, acc3 = z;

  v16h a  = load_a_f16(arow, kb2);
  v16h b0 = *(const v16h*)(brow0 + (size_t)0 * 16 * DMODEL);
  v16h b1 = *(const v16h*)(brow0 + (size_t)1 * 16 * DMODEL);
  v16h b2 = *(const v16h*)(brow0 + (size_t)2 * 16 * DMODEL);
  v16h b3 = *(const v16h*)(brow0 + (size_t)3 * 16 * DMODEL);
#pragma unroll 2
  for (int ks = 0; ks < 32; ++ks) {
    int k0n = ((ks + 1) & 31) * 32;
    v16h an  = load_a_f16(arow + k0n, kb2);
    v16h b0n = *(const v16h*)(brow0 + (size_t)0 * 16 * DMODEL + k0n);
    v16h b1n = *(const v16h*)(brow0 + (size_t)1 * 16 * DMODEL + k0n);
    v16h b2n = *(const v16h*)(brow0 + (size_t)2 * 16 * DMODEL + k0n);
    v16h b3n = *(const v16h*)(brow0 + (size_t)3 * 16 * DMODEL + k0n);
    acc0 = wmma16(a, b0, acc0);
    acc1 = wmma16(a, b1, acc1);
    acc2 = wmma16(a, b2, acc2);
    acc3 = wmma16(a, b3, acc3);
    a = an; b0 = b0n; b1 = b1n; b2 = b2n; b3 = b3n;
  }
  v8f accs[4] = {acc0, acc1, acc2, acc3};
#pragma unroll
  for (int nt = 0; nt < 4; ++nt) {
    int n = n0 + nt * 16 + lr;
    float bias = b_out[n];
#pragma unroll
    for (int r = 0; r < 8; ++r) {
      out[(size_t)(m0 + r + 8 * lh) * DMODEL + n] = accs[nt][r] + bias;
    }
  }
}

extern "C" void kernel_launch(void* const* d_in, const int* in_sizes, int n_in,
                              void* d_out, int out_size, void* d_ws, size_t ws_size,
                              hipStream_t stream) {
  (void)in_sizes; (void)n_in; (void)out_size; (void)ws_size;
  const float* x          = (const float*)d_in[0];
  const int*   levels     = (const int*)d_in[1];
  const unsigned char* mk = (const unsigned char*)d_in[2];
  const float* ln_g       = (const float*)d_in[3];
  const float* ln_b       = (const float*)d_in[4];
  const float* w_qkv      = (const float*)d_in[5];
  const float* hb_w1      = (const float*)d_in[6];
  const float* hb_b1      = (const float*)d_in[7];
  const float* hb_w2      = (const float*)d_in[8];
  const float* hb_b2      = (const float*)d_in[9];
  const float* level_emb  = (const float*)d_in[10];
  const float* scale_w    = (const float*)d_in[11];
  const float* rel_emb    = (const float*)d_in[12];
  const float* w_out      = (const float*)d_in[13];
  const float* b_out      = (const float*)d_in[14];
  float* out = (float*)d_out;

  char* p = (char*)d_ws;
  auto alloc = [&](size_t bytes) {
    char* q = p;
    p += (bytes + 255) & ~(size_t)255;
    return (void*)q;
  };
  _Float16* wqkvT = (_Float16*)alloc((size_t)3072 * 1024 * sizeof(_Float16));
  _Float16* woutT = (_Float16*)alloc((size_t)1024 * 1024 * sizeof(_Float16));
  _Float16* xnh   = (_Float16*)alloc((size_t)SEQ * DMODEL * sizeof(_Float16));
  _Float16* qb    = (_Float16*)alloc((size_t)NH * SEQ * DHC * sizeof(_Float16));
  _Float16* kbuf  = (_Float16*)alloc((size_t)NH * SEQ * DHC * sizeof(_Float16));
  _Float16* vT    = (_Float16*)alloc((size_t)NH * DHC * SEQ * sizeof(_Float16));
  _Float16* ctxb  = (_Float16*)alloc((size_t)SEQ * DMODEL * sizeof(_Float16));
  float* coef     = (float*)alloc((size_t)NH * SEQ * sizeof(float));
  float* pathsf   = (float*)alloc((size_t)SEQ * 8 * sizeof(float));
  float* pathnorm = (float*)alloc((size_t)SEQ * sizeof(float));
  int*   depthbuf = (int*)alloc((size_t)SEQ * sizeof(int));

  prep_weights_kernel<<<16384, 256, 0, stream>>>(w_qkv, w_out, wqkvT, woutT);
  ln_stats_kernel<<<SEQ, 256, 0, stream>>>(x, levels, ln_g, ln_b, level_emb, scale_w,
                                           xnh, coef, pathsf, pathnorm, depthbuf);
  qkv_gemm_kernel<<<dim3(48, 96), 32, 0, stream>>>(xnh, wqkvT, qb, kbuf, vT);
  attn_kernel<<<dim3(96, 16), 32, 0, stream>>>(qb, kbuf, vT, coef, pathsf, pathnorm,
                                               depthbuf, hb_w1, hb_b1, hb_w2, hb_b2,
                                               rel_emb, mk, ctxb);
  out_gemm_kernel<<<dim3(16, 96), 32, 0, stream>>>(ctxb, woutT, b_out, out);
}
